// LocalAttentionBlock_87076166960150
// MI455X (gfx1250) — matrix operand order repaired and checked by reading, not compiled, not run on hardware
//
#include <hip/hip_runtime.h>
#include <hip/hip_bf16.h>
#include <math.h>

// ---- problem constants (from the reference) ----
#define HIDDEN 2048
#define NHEADS 16
#define NKV    4
#define HD     128
#define INTER  5632
#define BATCH  2
#define SEQ    2048
#define MTOK   (BATCH * SEQ)   // 4096 rows

typedef __attribute__((ext_vector_type(16))) __bf16 v16bf;
typedef __attribute__((ext_vector_type(8)))  __bf16 v8bf;
typedef __attribute__((ext_vector_type(8)))  float  v8f;

// Launder a value through an empty asm so the optimizer cannot see it is a
// compile-time constant (prevents catastrophic full unrolling of K-loops
// while keeping compile-time tile strides as immediate offsets).
#define OPAQUE_I32(x) asm volatile("" : "+v"(x))

static __device__ __forceinline__ v8f wmma_bf(v16bf a, v16bf b, v8f c) {
  // D(16x16,f32) = A(16x32,bf16) * B(32x16,bf16) + C
  return __builtin_amdgcn_wmma_f32_16x16x32_bf16(false, a, false, b, (short)0, c,
                                                 false, false);
}

static __device__ __forceinline__ v8f vzero8f() {
  v8f z = {0.f, 0.f, 0.f, 0.f, 0.f, 0.f, 0.f, 0.f};
  return z;
}

// 16-bit A-matrix fragment (16x32): lane holds K = kb..kb+15
// with kb = (lane>>4)*16. Caller passes p = rowBase + k + (lane>>4)*16.
static __device__ __forceinline__ v16bf load_a16(const __bf16* p) {
  v8bf lo = *(const v8bf*)(p);
  v8bf hi = *(const v8bf*)(p + 8);
  return __builtin_shufflevector(lo, hi, 0, 1, 2, 3, 4, 5, 6, 7,
                                 8, 9, 10, 11, 12, 13, 14, 15);
}

// ---------------- weight downconvert ----------------
__global__ void f32_to_bf16(const float* __restrict__ src, __bf16* __restrict__ dst,
                            size_t n) {
  size_t i = (size_t)blockIdx.x * blockDim.x + threadIdx.x;
  size_t stride = (size_t)gridDim.x * blockDim.x;
  for (; i < n; i += stride) dst[i] = (__bf16)src[i];
}

// ---------------- RMSNorm (fp32 in -> bf16 out) ----------------
__global__ void rmsnorm_bf16(const float* __restrict__ x, const float* __restrict__ w,
                             __bf16* __restrict__ out, int D) {
  const int row = blockIdx.x;
  const int tid = threadIdx.x;
  const float* xr = x + (size_t)row * D;
  float ss = 0.f;
  for (int i = tid; i < D; i += 256) { float v = xr[i]; ss += v * v; }
#pragma unroll
  for (int m = 16; m >= 1; m >>= 1) ss += __shfl_xor(ss, m, 32);
  __shared__ float red[8];
  if ((tid & 31) == 0) red[tid >> 5] = ss;
  __syncthreads();
  float tot = 0.f;
#pragma unroll
  for (int j = 0; j < 8; ++j) tot += red[j];
  const float r = rsqrtf(tot / (float)D + 1e-6f);
  for (int i = tid; i < D; i += 256)
    out[(size_t)row * D + i] = (__bf16)(xr[i] * r * w[i]);
}

// ---------------- generic WMMA GEMM:  C[M,N] = A[M,K] * W[N,K]^T ----------------
// EPI 0: store bf16.  EPI 1: out_f32[i] = resid[i] + acc (resid may alias out).
// K compile-time -> tile strides fold into load imm offsets; loop bound is
// laundered so the loop stays rolled. Each wave: 32 rows x 64 cols, register
// double-buffered, peeled final K-step.
template <int EPI, int K>
__global__ void gemm_wmma(const __bf16* __restrict__ A, const __bf16* __restrict__ W,
                          void* __restrict__ outp, const float* __restrict__ resid,
                          int N) {
  const int lane = threadIdx.x & 31;
  const int wave = threadIdx.x >> 5;
  const int rlow = lane & 15;
  const int rsel = lane >> 4;
  const int tileM = blockIdx.y << 5;              // 32 rows per block
  const int n0 = blockIdx.x * 256 + wave * 64;    // 64 cols per wave

  const unsigned aOff = (unsigned)((tileM + rlow) * K + rsel * 16);
  const unsigned wOff = (unsigned)((n0 + rlow) * K + rsel * 16);

  v8f acc[2][4];
#pragma unroll
  for (int m = 0; m < 2; ++m)
#pragma unroll
    for (int t = 0; t < 4; ++t) acc[m][t] = vzero8f();

  // prologue: fragments for k = 0
  v16bf am[2], bt[4];
  am[0] = load_a16(A + aOff);
  am[1] = load_a16(A + aOff + 16 * K);
  bt[0] = *(const v16bf*)(W + wOff);
  bt[1] = *(const v16bf*)(W + wOff + 16 * K);
  bt[2] = *(const v16bf*)(W + wOff + 32 * K);
  bt[3] = *(const v16bf*)(W + wOff + 48 * K);

  int kLimit = K - 32;
  OPAQUE_I32(kLimit);  // keep the loop rolled
  for (int k = 0; k < kLimit; k += 32) {
    const int kn = k + 32;
    __builtin_prefetch(W + wOff + kn + 512, 0, 3);
    __builtin_prefetch(A + aOff + kn + 512, 0, 3);
    v16bf amn[2], btn[4];
    amn[0] = load_a16(A + aOff + kn);
    amn[1] = load_a16(A + aOff + 16 * K + kn);
    btn[0] = *(const v16bf*)(W + wOff + kn);
    btn[1] = *(const v16bf*)(W + wOff + 16 * K + kn);
    btn[2] = *(const v16bf*)(W + wOff + 32 * K + kn);
    btn[3] = *(const v16bf*)(W + wOff + 48 * K + kn);
#pragma unroll
    for (int m = 0; m < 2; ++m)
#pragma unroll
      for (int t = 0; t < 4; ++t) acc[m][t] = wmma_bf(am[m], bt[t], acc[m][t]);
#pragma unroll
    for (int m = 0; m < 2; ++m) am[m] = amn[m];
#pragma unroll
    for (int t = 0; t < 4; ++t) bt[t] = btn[t];
  }
  // peeled final K-step (fragments already resident)
#pragma unroll
  for (int m = 0; m < 2; ++m)
#pragma unroll
    for (int t = 0; t < 4; ++t) acc[m][t] = wmma_bf(am[m], bt[t], acc[m][t]);

#pragma unroll
  for (int m = 0; m < 2; ++m) {
    const int crow = tileM + m * 16 + rsel * 8;
#pragma unroll
    for (int r = 0; r < 8; ++r) {
      const size_t i0 = (size_t)(crow + r) * N + n0 + rlow;
      if (EPI == 0) {
        __bf16* o = (__bf16*)outp;
        o[i0]      = (__bf16)acc[m][0][r];
        o[i0 + 16] = (__bf16)acc[m][1][r];
        o[i0 + 32] = (__bf16)acc[m][2][r];
        o[i0 + 48] = (__bf16)acc[m][3][r];
      } else {
        float* o = (float*)outp;
        o[i0]      = resid[i0]      + acc[m][0][r];
        o[i0 + 16] = resid[i0 + 16] + acc[m][1][r];
        o[i0 + 32] = resid[i0 + 32] + acc[m][2][r];
        o[i0 + 48] = resid[i0 + 48] + acc[m][3][r];
      }
    }
  }
}

// ---------------- fused gate/up GEMM + SiLU*mul -> bf16 ----------------
template <int K>
__global__ void gateup_wmma(const __bf16* __restrict__ A, const __bf16* __restrict__ Wg,
                            const __bf16* __restrict__ Wu, __bf16* __restrict__ act,
                            int N) {
  const int lane = threadIdx.x & 31;
  const int wave = threadIdx.x >> 5;
  const int rlow = lane & 15;
  const int rsel = lane >> 4;
  const int tileM = blockIdx.y << 4;
  const int n0 = blockIdx.x * 256 + wave * 64;

  const unsigned aOff = (unsigned)((tileM + rlow) * K + rsel * 16);
  const unsigned wOff = (unsigned)((n0 + rlow) * K + rsel * 16);

  v8f accg[4], accu[4];
#pragma unroll
  for (int t = 0; t < 4; ++t) { accg[t] = vzero8f(); accu[t] = vzero8f(); }

  v16bf a, bg[4], bu[4];
  a = load_a16(A + aOff);
#pragma unroll
  for (int t = 0; t < 4; ++t) {
    bg[t] = *(const v16bf*)(Wg + wOff + t * 16 * K);
    bu[t] = *(const v16bf*)(Wu + wOff + t * 16 * K);
  }

  int kLimit = K - 32;
  OPAQUE_I32(kLimit);
  for (int k = 0; k < kLimit; k += 32) {
    const int kn = k + 32;
    __builtin_prefetch(Wg + wOff + kn + 512, 0, 3);
    __builtin_prefetch(Wu + wOff + kn + 512, 0, 3);
    v16bf an = load_a16(A + aOff + kn);
    v16bf bgn[4], bun[4];
#pragma unroll
    for (int t = 0; t < 4; ++t) {
      bgn[t] = *(const v16bf*)(Wg + wOff + t * 16 * K + kn);
      bun[t] = *(const v16bf*)(Wu + wOff + t * 16 * K + kn);
    }
#pragma unroll
    for (int t = 0; t < 4; ++t) accg[t] = wmma_bf(a, bg[t], accg[t]);
#pragma unroll
    for (int t = 0; t < 4; ++t) accu[t] = wmma_bf(a, bu[t], accu[t]);
    a = an;
#pragma unroll
    for (int t = 0; t < 4; ++t) { bg[t] = bgn[t]; bu[t] = bun[t]; }
  }
#pragma unroll
  for (int t = 0; t < 4; ++t) accg[t] = wmma_bf(a, bg[t], accg[t]);
#pragma unroll
  for (int t = 0; t < 4; ++t) accu[t] = wmma_bf(a, bu[t], accu[t]);

  const int crow = tileM + rsel * 8;
#pragma unroll
  for (int r = 0; r < 8; ++r) {
    const size_t i0 = (size_t)(crow + r) * N + n0 + rlow;
#pragma unroll
    for (int t = 0; t < 4; ++t) {
      const float g = accg[t][r];
      const float u = accu[t][r];
      act[i0 + t * 16] = (__bf16)((g / (1.f + __expf(-g))) * u);
    }
  }
}

// ---------------- RoPE + head transpose for q,k ----------------
__global__ void rope_transpose(const __bf16* __restrict__ qbuf,
                               const __bf16* __restrict__ kbuf,
                               __bf16* __restrict__ qt, __bf16* __restrict__ kt) {
  const int d = threadIdx.x;            // 0..127
  const int t = blockIdx.x;             // 0..T-1
  const int slot = blockIdx.y;          // 0..15 = q heads, 16..19 = k heads
  const int b = blockIdx.z;
  const bool isq = slot < NHEADS;
  const __bf16* src = isq
      ? qbuf + ((size_t)(b * SEQ + t) * HIDDEN + slot * HD)
      : kbuf + ((size_t)(b * SEQ + t) * (NKV * HD) + (slot - NHEADS) * HD);
  const float x = (float)src[d];
  const float other = (float)src[(d < 64) ? d + 64 : d - 64];
  const float rot = (d < 64) ? -other : other;
  const int dh = d & 63;
  const float invf = __powf(10000.0f, -(float)(2 * dh) / 128.0f);
  const float ang = (float)t * invf;
  const float o = x * __cosf(ang) + rot * __sinf(ang);
  __bf16* dst = isq
      ? qt + (((size_t)(b * NHEADS + slot) * SEQ + t) * HD + d)
      : kt + (((size_t)(b * NKV + (slot - NHEADS)) * SEQ + t) * HD + d);
  *dst = (__bf16)o;
}

// v [B*T, 512] -> vT [B, 4, 128, T]
__global__ void v_transpose(const __bf16* __restrict__ vbuf, __bf16* __restrict__ vT) {
  const int d = threadIdx.x;
  const int t = blockIdx.x;
  const int hk = blockIdx.y;
  const int b = blockIdx.z;
  const __bf16 v = vbuf[(size_t)(b * SEQ + t) * (NKV * HD) + hk * HD + d];
  vT[((size_t)(b * NKV + hk) * HD + d) * SEQ + t] = v;
}

// ---------------- flash attention ----------------
// 2 waves per block per (b, head, 16-row q tile). Both waves compute the SAME
// score tile + softmax (wave-local, no cross-wave comms) but each accumulates
// only its half of the head-dim output: 4 PV tiles = 32 acc VGPRs per wave.
// Each wave has a private P buffer in LDS, so the inner loop has NO barriers.
// Peak pressure ~176 VGPRs even with fully hoisted loads -> no spills.
__global__ void attn_wmma(const __bf16* __restrict__ qt, const __bf16* __restrict__ kt,
                          const __bf16* __restrict__ vT, __bf16* __restrict__ outp) {
  const int tid = threadIdx.x;
  const int lane = tid & 31;
  const int wave = tid >> 5;
  const int rlow = lane & 15;
  const int rsel = lane >> 4;
  const int q0 = blockIdx.x << 4;
  const int h = blockIdx.y;
  const int b = blockIdx.z;
  const int hk = h / (NHEADS / NKV);

  const __bf16* Q = qt + ((size_t)(b * NHEADS + h) * SEQ) * HD;
  const __bf16* Km = kt + ((size_t)(b * NKV + hk) * SEQ) * HD;
  const __bf16* Vm = vT + ((size_t)(b * NKV + hk) * HD) * SEQ;

  __shared__ __bf16 qlds[16 * HD];      // Q tile: 16 rows x 128 (4KB)
  __shared__ __bf16 plds[2][16 * 32];   // per-wave P tile (2 x 1KB)

  // cooperative Q stage: 64 lanes each copy bf16 cols 0..7 and 16..23 of a 32-col chunk
  {
    const int qr = tid & 15;
    const int qh = (tid >> 4) * 32;
    const __bf16* gsrc = Q + (size_t)(q0 + qr) * HD + qh;
    __bf16* ldst = &qlds[qr * HD + qh];
    *(v8bf*)(ldst) = *(const v8bf*)(gsrc);
    *(v8bf*)(ldst + 16) = *(const v8bf*)(gsrc + 16);
  }
  __syncthreads();  // only barrier in the kernel

  float mrun[8], lrun[8];
  v8f acc[4];  // 4 PV tiles: head-dim cols wave*64 .. wave*64+63
#pragma unroll
  for (int r = 0; r < 8; ++r) { mrun[r] = -3.0e38f; lrun[r] = 0.f; }
#pragma unroll
  for (int t = 0; t < 4; ++t) acc[t] = vzero8f();

  const float scale = 0.08838834764831845f;  // 1/sqrt(128)
  const int kend = q0 + 16;                  // causal bound (exclusive)

  const unsigned qOff = (unsigned)(rlow * HD + rsel * 16);  // A-frag base in qlds
  const unsigned kBase = (unsigned)(rlow * HD + rsel * 16);
  const unsigned vBase = (unsigned)((wave * 64 + rlow) * SEQ + rsel * 16);
  __bf16* myp = &plds[wave][0];

  for (int k0 = 0; k0 < kend; k0 += 32) {
    const bool sub1 = (k0 + 16) < kend;
    const unsigned ko0 = kBase + (unsigned)(k0 * HD);

    // score subtile 0 (keys k0..k0+15)
    v8f s0 = vzero8f(), s1 = vzero8f();
    {
      v16bf kb0 = *(const v16bf*)(Km + ko0);
      v16bf kb1 = *(const v16bf*)(Km + ko0 + 32);
      v16bf kb2 = *(const v16bf*)(Km + ko0 + 64);
      v16bf kb3 = *(const v16bf*)(Km + ko0 + 96);
      s0 = wmma_bf(load_a16(&qlds[qOff]),      kb0, s0);
      s0 = wmma_bf(load_a16(&qlds[qOff + 32]), kb1, s0);
      s0 = wmma_bf(load_a16(&qlds[qOff + 64]), kb2, s0);
      s0 = wmma_bf(load_a16(&qlds[qOff + 96]), kb3, s0);
    }
    if (sub1) {  // score subtile 1 (keys k0+16..k0+31)
      const unsigned ko1 = ko0 + 16 * HD;
      v16bf kb0 = *(const v16bf*)(Km + ko1);
      v16bf kb1 = *(const v16bf*)(Km + ko1 + 32);
      v16bf kb2 = *(const v16bf*)(Km + ko1 + 64);
      v16bf kb3 = *(const v16bf*)(Km + ko1 + 96);
      s1 = wmma_bf(load_a16(&qlds[qOff]),      kb0, s1);
      s1 = wmma_bf(load_a16(&qlds[qOff + 32]), kb1, s1);
      s1 = wmma_bf(load_a16(&qlds[qOff + 64]), kb2, s1);
      s1 = wmma_bf(load_a16(&qlds[qOff + 96]), kb3, s1);
    }

    // --- online softmax (identical in both waves); element (r+8*rsel, rlow) ---
#pragma unroll
    for (int r = 0; r < 8; ++r) {
      const int rowg = q0 + r + 8 * rsel;
      const int c0 = k0 + rlow;
      const int c1 = k0 + 16 + rlow;
      float a0 = (c0 <= rowg) ? s0[r] * scale : -3.0e38f;
      float a1 = (c1 <= rowg) ? s1[r] * scale : -3.0e38f;
      float mx = fmaxf(a0, a1);
      mx = fmaxf(mx, __shfl_xor(mx, 1, 32));
      mx = fmaxf(mx, __shfl_xor(mx, 2, 32));
      mx = fmaxf(mx, __shfl_xor(mx, 4, 32));
      mx = fmaxf(mx, __shfl_xor(mx, 8, 32));
      const float nm = fmaxf(mrun[r], mx);
      const float esc = __expf(mrun[r] - nm);
      const float p0 = __expf(a0 - nm);
      const float p1 = __expf(a1 - nm);
      float sm = p0 + p1;
      sm += __shfl_xor(sm, 1, 32);
      sm += __shfl_xor(sm, 2, 32);
      sm += __shfl_xor(sm, 4, 32);
      sm += __shfl_xor(sm, 8, 32);
      lrun[r] = lrun[r] * esc + sm;
      mrun[r] = nm;
#pragma unroll
      for (int t = 0; t < 4; ++t) acc[t][r] *= esc;
      const int lr = r + 8 * rsel;
      myp[lr * 32 + rlow] = (__bf16)p0;
      myp[lr * 32 + 16 + rlow] = (__bf16)p1;
    }

    // P (16x32) @ V-half (32x64): 4 WMMAs; no barrier (wave-private P buffer,
    // DS ops from the same wave are in-order)
    const unsigned vo = vBase + (unsigned)k0;
    v16bf vb0 = *(const v16bf*)(Vm + vo);
    v16bf vb1 = *(const v16bf*)(Vm + vo + 16 * SEQ);
    v16bf vb2 = *(const v16bf*)(Vm + vo + 32 * SEQ);
    v16bf vb3 = *(const v16bf*)(Vm + vo + 48 * SEQ);
    const v16bf pa = load_a16(myp + rlow * 32 + rsel * 16);
    acc[0] = wmma_bf(pa, vb0, acc[0]);
    acc[1] = wmma_bf(pa, vb1, acc[1]);
    acc[2] = wmma_bf(pa, vb2, acc[2]);
    acc[3] = wmma_bf(pa, vb3, acc[3]);
  }

  // normalize + write [B*T, 2048] bf16 (wave's 64-col half of the head)
#pragma unroll
  for (int r = 0; r < 8; ++r) {
    const float inv = 1.0f / lrun[r];
    const int tok = b * SEQ + q0 + r + 8 * rsel;
    __bf16* orow = outp + (size_t)tok * HIDDEN + h * HD + wave * 64;
#pragma unroll
    for (int t = 0; t < 4; ++t) orow[t * 16 + rlow] = (__bf16)(acc[t][r] * inv);
  }
}

// ======================= launcher =======================
extern "C" void kernel_launch(void* const* d_in, const int* in_sizes, int n_in,
                              void* d_out, int out_size, void* d_ws, size_t ws_size,
                              hipStream_t stream) {
  const float* x   = (const float*)d_in[0];
  const float* ln1 = (const float*)d_in[1];
  const float* ln2 = (const float*)d_in[2];
  const float* wq  = (const float*)d_in[3];
  const float* wk  = (const float*)d_in[4];
  const float* wv  = (const float*)d_in[5];
  const float* wo  = (const float*)d_in[6];
  const float* wg  = (const float*)d_in[7];
  const float* wu  = (const float*)d_in[8];
  const float* wd  = (const float*)d_in[9];
  float* out = (float*)d_out;

  char* ws = (char*)d_ws;
  size_t off = 0;
  auto alloc = [&](size_t elems) -> __bf16* {
    __bf16* p = (__bf16*)(ws + off);
    off = (off + elems * sizeof(__bf16) + 255) & ~(size_t)255;
    return p;
  };

  __bf16* wq_b = alloc((size_t)HIDDEN * HIDDEN);
  __bf16* wk_b = alloc((size_t)NKV * HD * HIDDEN);
  __bf16* wv_b = alloc((size_t)NKV * HD * HIDDEN);
  __bf16* wo_b = alloc((size_t)HIDDEN * HIDDEN);
  __bf16* wg_b = alloc((size_t)INTER * HIDDEN);
  __bf16* wu_b = alloc((size_t)INTER * HIDDEN);
  __bf16* wd_b = alloc((size_t)HIDDEN * INTER);
  __bf16* h_b  = alloc((size_t)MTOK * HIDDEN);
  __bf16* q_b  = alloc((size_t)MTOK * HIDDEN);
  __bf16* k_b  = alloc((size_t)MTOK * NKV * HD);
  __bf16* v_b  = alloc((size_t)MTOK * NKV * HD);
  __bf16* qt_b = alloc((size_t)MTOK * HIDDEN);
  __bf16* kt_b = alloc((size_t)MTOK * NKV * HD);
  __bf16* vT_b = alloc((size_t)MTOK * NKV * HD);
  __bf16* ao_b = alloc((size_t)MTOK * HIDDEN);
  __bf16* h2_b = alloc((size_t)MTOK * HIDDEN);
  __bf16* act_b = alloc((size_t)MTOK * INTER);

  // 1) weights -> bf16 (L2-resident afterwards)
  f32_to_bf16<<<512, 256, 0, stream>>>(wq, wq_b, (size_t)HIDDEN * HIDDEN);
  f32_to_bf16<<<512, 256, 0, stream>>>(wk, wk_b, (size_t)NKV * HD * HIDDEN);
  f32_to_bf16<<<512, 256, 0, stream>>>(wv, wv_b, (size_t)NKV * HD * HIDDEN);
  f32_to_bf16<<<512, 256, 0, stream>>>(wo, wo_b, (size_t)HIDDEN * HIDDEN);
  f32_to_bf16<<<512, 256, 0, stream>>>(wg, wg_b, (size_t)INTER * HIDDEN);
  f32_to_bf16<<<512, 256, 0, stream>>>(wu, wu_b, (size_t)INTER * HIDDEN);
  f32_to_bf16<<<512, 256, 0, stream>>>(wd, wd_b, (size_t)HIDDEN * INTER);

  const dim3 gblk(128, 1, 1);  // 4 waves per block (256 output cols)

  // 2) h = rmsnorm(x, ln1) -> bf16
  rmsnorm_bf16<<<MTOK, 256, 0, stream>>>(x, ln1, h_b, HIDDEN);

  // 3) q/k/v projections (WMMA)
  gemm_wmma<0, HIDDEN><<<dim3(HIDDEN / 256, MTOK / 32), gblk, 0, stream>>>(
      h_b, wq_b, q_b, nullptr, HIDDEN);
  gemm_wmma<0, HIDDEN><<<dim3((NKV * HD) / 256, MTOK / 32), gblk, 0, stream>>>(
      h_b, wk_b, k_b, nullptr, NKV * HD);
  gemm_wmma<0, HIDDEN><<<dim3((NKV * HD) / 256, MTOK / 32), gblk, 0, stream>>>(
      h_b, wv_b, v_b, nullptr, NKV * HD);

  // 4) RoPE + [B,H,T,D] transpose; V -> [B,Hkv,D,T]
  rope_transpose<<<dim3(SEQ, NHEADS + NKV, BATCH), 128, 0, stream>>>(q_b, k_b, qt_b, kt_b);
  v_transpose<<<dim3(SEQ, NKV, BATCH), 128, 0, stream>>>(v_b, vT_b);

  // 5) flash attention (WMMA QK^T + WMMA PV, online softmax); 2 waves/block
  attn_wmma<<<dim3(SEQ / 16, NHEADS, BATCH), 64, 0, stream>>>(qt_b, kt_b, vT_b, ao_b);

  // 6) x1 = x + attn @ wo^T  -> d_out (fp32)
  gemm_wmma<1, HIDDEN><<<dim3(HIDDEN / 256, MTOK / 32), gblk, 0, stream>>>(
      ao_b, wo_b, out, x, HIDDEN);

  // 7) h2 = rmsnorm(x1, ln2) -> bf16
  rmsnorm_bf16<<<MTOK, 256, 0, stream>>>(out, ln2, h2_b, HIDDEN);

  // 8) act = silu(h2@wg^T) * (h2@wu^T)  (fused, bf16 out)
  gateup_wmma<HIDDEN><<<dim3(INTER / 256, MTOK / 16), gblk, 0, stream>>>(
      h2_b, wg_b, wu_b, act_b, INTER);

  // 9) d_out = x1 + act @ wd^T  (in-place residual add)
  gemm_wmma<1, INTER><<<dim3(HIDDEN / 256, MTOK / 32), gblk, 0, stream>>>(
      act_b, wd_b, out, out, HIDDEN);
}
